// EATTS_44822278701146
// MI455X (gfx1250) — matrix operand order repaired and checked by reading, not compiled, not run on hardware
//
#include <hip/hip_runtime.h>
#include <stdint.h>

// Problem constants (from reference): B=2, N=1024, C=64
#define BB 2
#define NN 1024
#define CC 64
#define SLOPE 0.2f

typedef __attribute__((ext_vector_type(16))) _Float16 v16h;
typedef __attribute__((ext_vector_type(8)))  float    v8f;

union Frag16 { unsigned u[8]; v16h v; };

__device__ __forceinline__ float leaky(float x) { return x > 0.f ? x : SLOPE * x; }

// Two 16-byte loads -> one WMMA operand fragment (compiler emits *_load_b128).
__device__ __forceinline__ void ld2x128(Frag16& f, const void* p0, const void* p1) {
    uint4 a = *(const uint4*)p0;
    uint4 b = *(const uint4*)p1;
    f.u[0] = a.x; f.u[1] = a.y; f.u[2] = a.z; f.u[3] = a.w;
    f.u[4] = b.x; f.u[5] = b.y; f.u[6] = b.z; f.u[7] = b.w;
}

// ---------------------------------------------------------------------------
// Prep kernel: one block per (b, j) row.
//   iv[b,j,c] = X[b,j,:] @ W2[:,c]         (f32, workspace)
//   q[b,j]    = sum_c yita[c]*iv[b,j,c]    (rank-1 softmax precursor)
//   eT[b,c,j] = f16(X[b,j,c])              (transposed fp16 copy: WMMA B-frags
//                                           become contiguous b128 loads)
//   lWh       = f16(linW)                  (first 64 blocks)
// ---------------------------------------------------------------------------
__global__ __launch_bounds__(CC)
void k_prep(const float* __restrict__ X, const float* __restrict__ W2,
            const float* __restrict__ yita, const float* __restrict__ linW,
            float* __restrict__ iv, float* __restrict__ q,
            _Float16* __restrict__ eT, _Float16* __restrict__ lWh)
{
    __shared__ float row_s[CC];
    __shared__ float red_s[CC];
    const int c  = threadIdx.x;           // 0..63
    const int bj = blockIdx.x;            // b*N + j
    const int b  = bj >> 10, j = bj & (NN - 1);

    float a = X[(size_t)bj * CC + c];
    row_s[c] = a;
    eT[((size_t)b * CC + c) * NN + j] = (_Float16)a;   // transpose to [b][c][j]
    __syncthreads();

    float acc = 0.f;
#pragma unroll 8
    for (int k = 0; k < CC; ++k) acc = fmaf(row_s[k], W2[k * CC + c], acc);
    iv[(size_t)bj * CC + c] = acc;

    red_s[c] = yita[c] * acc;
    __syncthreads();
    for (int s = 32; s > 0; s >>= 1) {
        if (c < s) red_s[c] += red_s[c + s];
        __syncthreads();
    }
    if (c == 0) q[bj] = red_s[0];

    if (bj < CC) lWh[bj * CC + c] = (_Float16)linW[bj * CC + c];
}

// ---------------------------------------------------------------------------
// Main fused kernel: one block (256 thr = 8 wave32) per (b, 16-row i-tile).
//   phase1: vu tile + p[i]                     (tiny VALU gemm)
//   phase2: scores = 0.4*sum yita|vu+iv| + 0.6*(p[i]+q[j]), adj mask
//   phase3: in-LDS row softmax; alpha written as fp16 in A-frag-friendly layout
//   phase4: agg = alpha @ X   v_wmma_f32_16x16x32_f16, K split over wave pairs
//   phase5: x = leaky(agg) + (1+beta)*X  -> fp16 LDS
//   phase6: out = leaky(x @ linW^T + b)  second fp16 WMMA (K=64)
// ---------------------------------------------------------------------------
__global__ __launch_bounds__(256)
void k_gat(const float* __restrict__ X, const int* __restrict__ adj,
           const float* __restrict__ U2, const float* __restrict__ yita,
           const float* __restrict__ beta, const float* __restrict__ linb,
           const float* __restrict__ iv, const float* __restrict__ q,
           const _Float16* __restrict__ eT, const _Float16* __restrict__ lWh,
           float* __restrict__ out)
{
    __shared__ float attr_s[16][68];                    // X tile
    __shared__ float vu_s[16][68];                      // vu tile
    __shared__ float sc_s[16][1028];                    // raw scores (f32)
    __shared__ __align__(16) _Float16 a_s[16][1032];    // alpha fp16 (516-dword stride: conflict-free b128)
    __shared__ __align__(16) _Float16 x_s[16][72];      // residual-mix fp16
    __shared__ float ps[4][16][17];                     // partial-D reduction
    __shared__ float p_s[16];
    __shared__ float y_s[CC];                           // 0.4*yita

    const int tid  = threadIdx.x;
    const int lane = tid & 31;
    const int wv   = tid >> 5;
    const int b    = blockIdx.y;
    const int i0   = blockIdx.x * 16;

    // ---- phase 0: load X tile + scaled yita ----
    for (int r = 0; r < 4; ++r) {
        int idx = tid + r * 256;
        int i = idx >> 6, c = idx & 63;
        attr_s[i][c] = X[((size_t)(b * NN + i0 + i)) * CC + c];
    }
    if (tid < CC) y_s[tid] = 0.4f * yita[tid];
    __syncthreads();

    // ---- phase 1: vu tile + p[i] ----
    for (int r = 0; r < 4; ++r) {
        int idx = tid + r * 256;
        int i = idx >> 6, c = idx & 63;
        float acc = 0.f;
#pragma unroll 8
        for (int k = 0; k < CC; ++k) acc = fmaf(attr_s[i][k], U2[k * CC + c], acc);
        vu_s[i][c] = acc;
    }
    __syncthreads();
    if (tid < 16) {
        float acc = 0.f;
        for (int c = 0; c < CC; ++c) acc = fmaf(yita[c], vu_s[tid][c], acc);
        p_s[tid] = acc;
    }
    __syncthreads();

    // ---- phase 2: masked scores (leaky split: 0.6*linear + 0.4*|.|) ----
    for (int r = 0; r < 4; ++r) {
        int j = tid + r * 256;
        float acc[16];
#pragma unroll
        for (int i = 0; i < 16; ++i) acc[i] = 0.f;
        const float* ivj = iv + ((size_t)(b * NN + j)) * CC;
        for (int c = 0; c < CC; ++c) {
            float ivc = ivj[c];
            float yc  = y_s[c];
#pragma unroll
            for (int i = 0; i < 16; ++i)
                acc[i] = fmaf(yc, fabsf(vu_s[i][c] + ivc), acc[i]);
        }
        float qj = q[b * NN + j];
#pragma unroll
        for (int i = 0; i < 16; ++i) {
            float s = acc[i] + 0.6f * (p_s[i] + qj);
            int a = adj[(((size_t)b * NN) + i0 + i) * NN + j];
            sc_s[i][j] = (a > 0) ? s : -1e12f;
        }
    }
    __syncthreads();

    // ---- phase 3: softmax over j; write normalized alpha as fp16 ----
    for (int rr = 0; rr < 2; ++rr) {
        int i = wv * 2 + rr;
        float m = -3.4e38f;
        for (int j = lane; j < NN; j += 32) m = fmaxf(m, sc_s[i][j]);
        for (int off = 16; off > 0; off >>= 1) m = fmaxf(m, __shfl_xor(m, off, 32));
        float s = 0.f;
        for (int j = lane; j < NN; j += 32) {
            float e = __expf(sc_s[i][j] - m);
            sc_s[i][j] = e;
            s += e;
        }
        for (int off = 16; off > 0; off >>= 1) s += __shfl_xor(s, off, 32);
        float inv = 1.f / s;
        for (int j = lane; j < NN; j += 32) a_s[i][j] = (_Float16)(sc_s[i][j] * inv);
    }
    __syncthreads();

    const int hi    = lane >> 4;      // K-half select in A/B fragment layouts
    const int lo16  = lane & 15;
    const int ctile = wv & 3;         // 16-col output tile
    const int kh    = wv >> 2;        // K-half of the 1024-deep reduction

    // ---- phase 4: agg GEMM alpha[16x1024] @ X_f16[1024x64] (all 8 waves) ----
    v8f d = {0.f, 0.f, 0.f, 0.f, 0.f, 0.f, 0.f, 0.f};
    {
        const _Float16* eTc  = eT + (((size_t)b * CC) + ctile * 16 + lo16) * NN;
        const _Float16* arow = &a_s[lo16][0];
        const int kbeg = kh * 512, kend = kbeg + 512;
#pragma unroll 2
        for (int k0 = kbeg; k0 < kend; k0 += 32) {
            Frag16 A, Bm;
            // A 16x32 f16: lanes' V0-3 = K [k0+8hi, +8), V4-7 = K [k0+16+8hi, +8)
            ld2x128(A, arow + k0 + 8 * hi, arow + k0 + 16 + 8 * hi);
            // B 32x16 f16: V0-3 = K [k0+8hi, +8), V4-7 = K [k0+16+8hi, +8)
            ld2x128(Bm, eTc + k0 + 8 * hi, eTc + k0 + 16 + 8 * hi);
            __builtin_prefetch((const void*)(eTc + k0 + 256), 0, 0);
            d = __builtin_amdgcn_wmma_f32_16x16x32_f16(false, A.v, false, Bm.v,
                                                       (short)0, d, false, false);
        }
    }
    if (wv >= 4) {                    // second K-half: spill partials
#pragma unroll
        for (int rf = 0; rf < 8; ++rf) ps[ctile][rf + 8 * hi][lo16] = d[rf];
    }
    __syncthreads();

    // ---- phase 5: reduce partials, residual mix -> fp16 staging ----
    if (wv < 4) {
#pragma unroll
        for (int rf = 0; rf < 8; ++rf) d[rf] += ps[ctile][rf + 8 * hi][lo16];
        float bet = 1.f + beta[0];
        int c = ctile * 16 + lo16;
#pragma unroll
        for (int rf = 0; rf < 8; ++rf) {   // D: m = rf + 8*hi, n = lo16
            int i = rf + 8 * hi;
            x_s[i][c] = (_Float16)(leaky(d[rf]) + bet * attr_s[i][c]);
        }
    }
    __syncthreads();

    // ---- phase 6: out = leaky(x @ linW^T + b) ----
    if (wv < 4) {
        const int o = ctile * 16 + lo16;
        v8f d2 = {0.f, 0.f, 0.f, 0.f, 0.f, 0.f, 0.f, 0.f};
        const _Float16* wrow = lWh + (size_t)o * CC;
#pragma unroll
        for (int k0 = 0; k0 < CC; k0 += 32) {
            Frag16 A, Bm;
            ld2x128(A, &x_s[lo16][k0 + 8 * hi], &x_s[lo16][k0 + 16 + 8 * hi]);
            ld2x128(Bm, wrow + k0 + 8 * hi, wrow + k0 + 16 + 8 * hi);
            d2 = __builtin_amdgcn_wmma_f32_16x16x32_f16(false, A.v, false, Bm.v,
                                                        (short)0, d2, false, false);
        }
        float bo = linb[o];
#pragma unroll
        for (int rf = 0; rf < 8; ++rf) {
            int i = rf + 8 * hi;
            out[(((size_t)b * NN) + i0 + i) * CC + o] = leaky(d2[rf] + bo);
        }
    }
}

// ---------------------------------------------------------------------------
// Workspace layout (bytes):
//   iv   f32 [B*N*C]  @ 0        (524288)
//   q    f32 [B*N]    @ 524288   (8192)
//   eT   f16 [B*C*N]  @ 532480   (262144)
//   lWh  f16 [C*C]    @ 794624   (8192)     total ~785 KB
// ---------------------------------------------------------------------------
extern "C" void kernel_launch(void* const* d_in, const int* in_sizes, int n_in,
                              void* d_out, int out_size, void* d_ws, size_t ws_size,
                              hipStream_t stream)
{
    const float* edge_attr = (const float*)d_in[0];
    const int*   edge_adj  = (const int*)d_in[1];
    // d_in[2] = e_max (unused)
    const float* U2   = (const float*)d_in[3];
    const float* W2   = (const float*)d_in[4];
    const float* yita = (const float*)d_in[5];
    const float* beta = (const float*)d_in[6];
    const float* linW = (const float*)d_in[7];
    const float* linb = (const float*)d_in[8];
    float* out = (float*)d_out;

    char* ws = (char*)d_ws;
    float*     iv  = (float*)(ws);
    float*     q   = (float*)(ws + 524288);
    _Float16*  eT  = (_Float16*)(ws + 532480);
    _Float16*  lWh = (_Float16*)(ws + 794624);

    k_prep<<<dim3(BB * NN), dim3(CC), 0, stream>>>(edge_attr, W2, yita, linW,
                                                   iv, q, eT, lWh);
    k_gat<<<dim3(NN / 16, BB), dim3(256), 0, stream>>>(edge_attr, edge_adj, U2, yita,
                                                       beta, linb, iv, q, eT, lWh, out);
}
